// RMSNormLSTM_75462575390836
// MI455X (gfx1250) — matrix operand order repaired and checked by reading, not compiled, not run on hardware
//
#include <hip/hip_runtime.h>
#include <hip/hip_bf16.h>
#include <stdint.h>

// Problem constants (match reference)
#define SEQ   512
#define BATCH 64
#define IN    512
#define HID   1024
#define GATES (4 * HID)        // 4096
#define ROWS  (SEQ * BATCH)    // 32768
#define EPSV  1e-5f

typedef __attribute__((ext_vector_type(16))) __bf16         v16bf;
typedef __attribute__((ext_vector_type(8)))  float          v8f;
typedef __attribute__((ext_vector_type(8)))  unsigned short us8;
typedef __attribute__((ext_vector_type(16))) unsigned short us16;

union Frag {
  us16  u;
  us8   h[2];
  v16bf bf;
};

__device__ __forceinline__ unsigned short f2bf(float f) {
  unsigned int u = __float_as_uint(f);
  u += 0x7FFFu + ((u >> 16) & 1u);   // round-to-nearest-even
  return (unsigned short)(u >> 16);
}

// ---------------- conversion / init kernels ----------------
__global__ void k_f32_to_bf16(const float* __restrict__ src,
                              unsigned short* __restrict__ dst, int n) {
  int i = blockIdx.x * blockDim.x + threadIdx.x;
  int stride = gridDim.x * blockDim.x;
  for (; i < n; i += stride) dst[i] = f2bf(src[i]);
}

__global__ void k_zero_state(float* __restrict__ c, float* __restrict__ h,
                             unsigned short* __restrict__ hb) {
  int i = blockIdx.x * blockDim.x + threadIdx.x;  // covers BATCH*HID
  c[i] = 0.f;
  h[i] = 0.f;
  hb[i] = 0;
}

// ---------------- WMMA GEMM: C = A(MxK,bf16) * W(NxK,bf16)^T + bias (+ addend) -------------
// Block: 256 threads = 8 waves as 2(M) x 4(N). Wave tile = 32 x (NI*16).
//   NI=4: block tile 64x256, 8 WMMA per wave per K-step, 1.5 ds_loads/WMMA (big GEMM)
//   NI=2: block tile 64x128, wider grid dispersion (latency-critical recurrent step GEMM)
// A tile staged in LDS with K contiguous (no permutation), so each lane's A fragment
// (contiguous k = 16*(lane/16) + s: lanes 0-15 K0-15, lanes 16-31 K16-31) is one
// contiguous 32-byte read. B fragments (lane = n column, K contiguous) need no permute.
#define LDSA_S 40
#define LDSB_S 40

template <int NI>
__global__ __launch_bounds__(256)
void k_wmma_gemm(const unsigned short* __restrict__ A,  // M x K (row-major bf16)
                 const unsigned short* __restrict__ W,  // N x K (row-major bf16)
                 const float* __restrict__ bias,        // N
                 const float* __restrict__ addend,      // M x ldc or nullptr
                 float* __restrict__ C,                 // M x ldc
                 int K, int lda, int ldw, int ldc) {
  constexpr int BN = 64 * NI;          // block N tile (rows of W staged)
  __shared__ unsigned short lA[64 * LDSA_S];
  __shared__ unsigned short lB[BN * LDSB_S];

  const int tid  = threadIdx.x;
  const int wid  = tid >> 5;
  const int lane = tid & 31;
  const int lm   = lane & 15;
  const int lh   = lane >> 4;

  const int m0 = blockIdx.x * 64;
  const int n0 = blockIdx.y * BN;
  const int waveM = (wid >> 2) * 32;          // 0 / 32
  const int waveN = (wid & 3) * (NI * 16);    // wave's N offset in block tile

  // A cooperative-load coordinates: 64 rows x 4 K-chunks = 256 segments (1/thread)
  const int arow   = tid >> 2;
  const int achunk = tid & 3;
  const int apos   = achunk * 8;  // contiguous pos
  const unsigned short* aSrc = A + (size_t)(m0 + arow) * lda + achunk * 8;

  v8f acc[2][NI] = {};

  for (int k0 = 0; k0 < K; k0 += 32) {
    // ---- stage A tile (64 x 32) ----
    us8 av = *(const us8*)(aSrc + k0);
    if (k0 + 32 < K)
      __builtin_prefetch(aSrc + k0 + 32, 0, 3);   // global_prefetch_b8, near scope
    *(us8*)&lA[arow * LDSA_S + apos] = av;

    // ---- stage B tile (BN x 32): 32B segments, BN*2 of them ----
#pragma unroll
    for (int s = 0; s < NI / 2; ++s) {
      const int seg = tid + s * 256;
      const int row = seg >> 1;
      const int col = (seg & 1) * 16;
      const unsigned short* src = W + (size_t)(n0 + row) * ldw + col + k0;
      us8 b0 = *(const us8*)src;
      us8 b1 = *(const us8*)(src + 8);
      if (k0 + 32 < K) __builtin_prefetch(src + 32, 0, 3);
      *(us8*)&lB[row * LDSB_S + col]     = b0;
      *(us8*)&lB[row * LDSB_S + col + 8] = b1;
    }
    __syncthreads();

    // ---- fragments ----
    Frag af[2], bf[NI];
#pragma unroll
    for (int mi = 0; mi < 2; ++mi) {
      const unsigned short* p = &lA[(waveM + mi * 16 + lm) * LDSA_S + lh * 16];
      af[mi].h[0] = *(const us8*)p;
      af[mi].h[1] = *(const us8*)(p + 8);
    }
#pragma unroll
    for (int ni = 0; ni < NI; ++ni) {
      const unsigned short* p = &lB[(waveN + ni * 16 + lm) * LDSB_S + lh * 16];
      bf[ni].h[0] = *(const us8*)p;
      bf[ni].h[1] = *(const us8*)(p + 8);
    }

    // ---- matrix cores: 2 x NI WMMAs, A fragments reused across NI B fragments ----
#pragma unroll
    for (int mi = 0; mi < 2; ++mi)
#pragma unroll
      for (int ni = 0; ni < NI; ++ni)
        acc[mi][ni] = __builtin_amdgcn_wmma_f32_16x16x32_bf16(
            false, af[mi].bf, false, bf[ni].bf, (short)0, acc[mi][ni],
            false, false);
    __syncthreads();
  }

  // epilogue: C/D layout — VGPR r: lanes 0-15 -> M=r, lanes 16-31 -> M=8+r; N = lane%16
#pragma unroll
  for (int mi = 0; mi < 2; ++mi) {
#pragma unroll
    for (int ni = 0; ni < NI; ++ni) {
      const int n = n0 + waveN + ni * 16 + lm;
      const float bval = bias[n];
#pragma unroll
      for (int r = 0; r < 8; ++r) {
        const int m = m0 + waveM + mi * 16 + lh * 8 + r;
        size_t idx = (size_t)m * ldc + n;
        float v = acc[mi][ni][r] + bval;
        if (addend) v += addend[idx];
        C[idx] = v;
      }
    }
  }
}

// ---------------- pointwise: RMSNorm gates, state update ----------------
__device__ __forceinline__ float blockSum(float v, float* sred) {
#pragma unroll
  for (int off = 16; off > 0; off >>= 1) v += __shfl_xor(v, off, 32);
  const int wid = threadIdx.x >> 5;
  __syncthreads();
  if ((threadIdx.x & 31) == 0) sred[wid] = v;
  __syncthreads();
  float t = 0.f;
#pragma unroll
  for (int j = 0; j < 8; ++j) t += sred[j];
  return t;
}

__device__ __forceinline__ float hsig(float x) {
  return fminf(fmaxf(0.2f * x + 0.5f, 0.f), 1.f);
}

__global__ __launch_bounds__(256)
void k_lstm_pointwise(const float* __restrict__ gates,  // BATCH x GATES
                      const float* __restrict__ w_i, const float* __restrict__ w_f,
                      const float* __restrict__ w_g, const float* __restrict__ w_o,
                      const float* __restrict__ w_c,
                      float* __restrict__ cst,               // BATCH x HID (state)
                      float* __restrict__ hst,               // BATCH x HID (state, f32)
                      float* __restrict__ h_out,             // nullable: hs output slice
                      unsigned short* __restrict__ h_out_bf, // nullable: next-layer input
                      unsigned short* __restrict__ h_bf) {   // BATCH x HID (state, bf16)
  __shared__ float sred[8];
  const int b = blockIdx.x;
  const int tid = threadIdx.x;
  const float* g = gates + (size_t)b * GATES;

  float xi[4], xf[4], xg[4], xo[4];
  float ssi = 0, ssf = 0, ssg = 0, sso = 0;
#pragma unroll
  for (int j = 0; j < 4; ++j) {
    int e = tid + j * 256;
    xi[j] = g[e];
    xf[j] = g[HID + e];
    xg[j] = g[2 * HID + e];
    xo[j] = g[3 * HID + e];
    ssi += xi[j] * xi[j]; ssf += xf[j] * xf[j];
    ssg += xg[j] * xg[j]; sso += xo[j] * xo[j];
  }
  const float inv = 1.f / (float)HID;
  float ri = rsqrtf(blockSum(ssi, sred) * inv + EPSV);
  float rf = rsqrtf(blockSum(ssf, sred) * inv + EPSV);
  float rg = rsqrtf(blockSum(ssg, sred) * inv + EPSV);
  float ro = rsqrtf(blockSum(sso, sred) * inv + EPSV);

  float cc[4], ov[4], ssc = 0.f;
#pragma unroll
  for (int j = 0; j < 4; ++j) {
    int e = tid + j * 256;
    float iv = hsig(xi[j] * ri * w_i[e]);
    float fv = hsig(xf[j] * rf * w_f[e]);
    float gn = xg[j] * rg * w_g[e];
    float gv = gn / (1.f + expf(-gn));           // swish
    ov[j] = hsig(xo[j] * ro * w_o[e]);
    size_t idx = (size_t)b * HID + e;
    float cold = cst[idx];
    cc[j] = fv * cold + iv * gv;
    cst[idx] = cc[j];
    ssc += cc[j] * cc[j];
  }
  float rc = rsqrtf(blockSum(ssc, sred) * inv + EPSV);
#pragma unroll
  for (int j = 0; j < 4; ++j) {
    int e = tid + j * 256;
    size_t idx = (size_t)b * HID + e;
    float hn = ov[j] * tanhf(cc[j] * rc * w_c[e]);
    hst[idx] = hn;
    unsigned short hb16 = f2bf(hn);
    h_bf[idx] = hb16;
    if (h_out) h_out[idx] = hn;
    if (h_out_bf) h_out_bf[idx] = hb16;
  }
}

// ---------------- host ----------------
extern "C" void kernel_launch(void* const* d_in, const int* in_sizes, int n_in,
                              void* d_out, int out_size, void* d_ws, size_t ws_size,
                              hipStream_t stream) {
  const float* x     = (const float*)d_in[0];
  const float* Wx[2] = {(const float*)d_in[1], (const float*)d_in[10]};
  const float* bx[2] = {(const float*)d_in[2], (const float*)d_in[11]};
  const float* Wh[2] = {(const float*)d_in[3], (const float*)d_in[12]};
  const float* bh[2] = {(const float*)d_in[4], (const float*)d_in[13]};
  const float* wi[2] = {(const float*)d_in[5], (const float*)d_in[14]};
  const float* wf[2] = {(const float*)d_in[6], (const float*)d_in[15]};
  const float* wg[2] = {(const float*)d_in[7], (const float*)d_in[16]};
  const float* wo[2] = {(const float*)d_in[8], (const float*)d_in[17]};
  const float* wc[2] = {(const float*)d_in[9], (const float*)d_in[18]};

  // workspace carve-up
  char* p = (char*)d_ws;
  float* xg = (float*)p;                    p += (size_t)ROWS * GATES * 4;  // 512 MB
  unsigned short* act = (unsigned short*)p; p += (size_t)ROWS * HID * 2;    // 64 MB (x_bf16 then hs0_bf16)
  unsigned short* wxb[2];
  unsigned short* whb[2];
  wxb[0] = (unsigned short*)p; p += (size_t)GATES * IN * 2;
  whb[0] = (unsigned short*)p; p += (size_t)GATES * HID * 2;
  wxb[1] = (unsigned short*)p; p += (size_t)GATES * HID * 2;
  whb[1] = (unsigned short*)p; p += (size_t)GATES * HID * 2;
  float* gates = (float*)p;                 p += (size_t)BATCH * GATES * 4;
  float* cf = (float*)p;                    p += (size_t)BATCH * HID * 4;
  float* hf = (float*)p;                    p += (size_t)BATCH * HID * 4;
  unsigned short* hb = (unsigned short*)p;  p += (size_t)BATCH * HID * 2;

  // fp32 -> bf16 conversions (bf16 weights ~28 MB: L2-resident afterwards)
  k_f32_to_bf16<<<2048, 256, 0, stream>>>(x, act, ROWS * IN);
  k_f32_to_bf16<<<1024, 256, 0, stream>>>(Wx[0], wxb[0], GATES * IN);
  k_f32_to_bf16<<<1024, 256, 0, stream>>>(Wh[0], whb[0], GATES * HID);
  k_f32_to_bf16<<<1024, 256, 0, stream>>>(Wx[1], wxb[1], GATES * HID);
  k_f32_to_bf16<<<1024, 256, 0, stream>>>(Wh[1], whb[1], GATES * HID);

  float* outp = (float*)d_out;
  const size_t HN_OFF = (size_t)SEQ * BATCH * HID;
  const size_t CN_OFF = HN_OFF + (size_t)2 * BATCH * HID;

  for (int l = 0; l < 2; ++l) {
    const int K = l ? HID : IN;
    // xg = act(ROWS x K) * Wx^T + bx : big parallel GEMM, fat 64x256 tile
    dim3 gg(ROWS / 64, GATES / 256);
    k_wmma_gemm<4><<<gg, 256, 0, stream>>>(act, wxb[l], bx[l], nullptr, xg,
                                           K, K, K, GATES);
    k_zero_state<<<(BATCH * HID) / 256, 256, 0, stream>>>(cf, hf, hb);

    for (int t = 0; t < SEQ; ++t) {
      // gates = xg[t] + h * Wh^T + bh : latency-critical, spread over 32 WGPs
      dim3 gs(1, GATES / 128);
      k_wmma_gemm<2><<<gs, 256, 0, stream>>>(hb, whb[l], bh[l],
                                             xg + (size_t)t * BATCH * GATES,
                                             gates, HID, HID, HID, GATES);
      float* hout = l ? outp + (size_t)t * BATCH * HID : nullptr;
      unsigned short* hbout = l ? nullptr : act + (size_t)t * BATCH * HID;
      k_lstm_pointwise<<<BATCH, 256, 0, stream>>>(gates, wi[l], wf[l], wg[l],
                                                  wo[l], wc[l], cf, hf, hout,
                                                  hbout, hb);
    }
    hipMemcpyAsync(outp + HN_OFF + (size_t)l * BATCH * HID, hf,
                   (size_t)BATCH * HID * sizeof(float),
                   hipMemcpyDeviceToDevice, stream);
    hipMemcpyAsync(outp + CN_OFF + (size_t)l * BATCH * HID, cf,
                   (size_t)BATCH * HID * sizeof(float),
                   hipMemcpyDeviceToDevice, stream);
  }
  (void)in_sizes; (void)n_in; (void)out_size; (void)ws_size;
}
